// StandardMHA_91079076479236
// MI455X (gfx1250) — matrix operand order repaired and checked by reading, not compiled, not run on hardware
//
#include <hip/hip_runtime.h>
#include <hip/hip_bf16.h>

typedef _Float16 half_t;
typedef __attribute__((ext_vector_type(16))) _Float16 v16h;
typedef __attribute__((ext_vector_type(8)))  _Float16 v8h;
typedef __attribute__((ext_vector_type(8)))  float    v8f;

union Frag16 { v16h v; v8h h[2]; };

// ---------------------------------------------------------------------------
// Fragment loaders per CDNA5 ISA layouts (CDNA5 ISA §7.12.2)
// A 16x32 f16 (row-major src, leading dim ld): lane r=lane&15 holds row r,
// element pairs: k = 8*half + 0..7 then k = 16 + 8*half + 0..7.
__device__ __forceinline__ v16h load_frag_a(const half_t* base, int ld) {
  const int lane = threadIdx.x & 31;
  const int r = lane & 15;
  const int h = lane >> 4;
  const half_t* p = base + r * ld + 8 * h;
  Frag16 f;
  f.h[0] = *(const v8h*)(p);       // k = 8h .. 8h+7
  f.h[1] = *(const v8h*)(p + 16);  // k = 16+8h .. 16+8h+7
  return f.v;
}

// B 32x16 f16 from [N][K]-layout source (leading dim ld): lane n=lane&15 holds
// column n; k = 8*half + 0..7 then k = 16 + 8*half + 0..7 (matches A).
__device__ __forceinline__ v16h load_frag_b(const half_t* base, int ld) {
  const int lane = threadIdx.x & 31;
  const int n = lane & 15;
  const int h = lane >> 4;
  const half_t* p = base + n * ld + 8 * h;
  Frag16 f;
  f.h[0] = *(const v8h*)(p);
  f.h[1] = *(const v8h*)(p + 16);
  return f.v;
}

__device__ __forceinline__ v8f wmma_f32(v16h a, v16h b, v8f c) {
  return __builtin_amdgcn_wmma_f32_16x16x32_f16(
      /*neg_a=*/false, a, /*neg_b=*/false, b,
      /*c_mod=*/(short)0, c, /*reuse_a=*/false, /*reuse_b=*/false);
}

// ---------------------------------------------------------------------------
// Elementwise f32 -> f16
__global__ __launch_bounds__(256) void cvt_f32_f16(const float* __restrict__ in,
                                                   half_t* __restrict__ out, int n) {
  int i = blockIdx.x * 256 + threadIdx.x;
  if (i < n) out[i] = (half_t)in[i];
}

// Transpose 1024x1024 f32 weight -> f16 [N][K] layout: wt[n*1024+k] = w[k*1024+n]
__global__ __launch_bounds__(256) void cvt_transpose(const float* __restrict__ w,
                                                     half_t* __restrict__ wt) {
  int i = blockIdx.x * 256 + threadIdx.x;  // i = n*1024 + k
  int n = i >> 10;
  int k = i & 1023;
  wt[i] = (half_t)w[(size_t)k * 1024 + n];
}

// ---------------------------------------------------------------------------
// QKV projection: out16 = (x16 @ W^T + b) * scale, stored per-head.
// 32x32 output tile per wave (2x2 WMMA tiles, A/B fragments reused 2x each).
// mode 0/1: [H][L][64] row-major (Q, K).  mode 2: [H][64][L] transposed (V).
__global__ __launch_bounds__(128) void qkv_proj_kernel(
    const half_t* __restrict__ x16, const half_t* __restrict__ wt,
    const float* __restrict__ bias, half_t* __restrict__ out,
    int mode, float scale) {
  const int wave = threadIdx.x >> 5;
  const int gid  = blockIdx.x * 4 + wave;      // 4096 waves
  const int m0   = (gid >> 5) * 32;            // 128 row tiles
  const int n0   = (gid & 31) * 32;            // 32 col tiles
  const int lane = threadIdx.x & 31;
  const int col  = lane & 15, h = lane >> 4;

  v8f acc[2][2] = {};
  const half_t* arow0 = x16 + (size_t)m0 * 1024;
  const half_t* arow1 = x16 + (size_t)(m0 + 16) * 1024;
  const half_t* brow0 = wt + (size_t)n0 * 1024;
  const half_t* brow1 = wt + (size_t)(n0 + 16) * 1024;
  for (int k = 0; k < 1024; k += 32) {
    v16h a0 = load_frag_a(arow0 + k, 1024);
    v16h a1 = load_frag_a(arow1 + k, 1024);
    v16h b0 = load_frag_b(brow0 + k, 1024);
    v16h b1 = load_frag_b(brow1 + k, 1024);
    acc[0][0] = wmma_f32(a0, b0, acc[0][0]);
    acc[0][1] = wmma_f32(a0, b1, acc[0][1]);
    acc[1][0] = wmma_f32(a1, b0, acc[1][0]);
    acc[1][1] = wmma_f32(a1, b1, acc[1][1]);
  }
#pragma unroll
  for (int ns = 0; ns < 2; ++ns) {
    const int nb   = n0 + 16 * ns;
    const float bb = bias[nb + col];
    const int head = nb >> 6;
    const int dcol = (nb & 63) + col;
#pragma unroll
    for (int qs = 0; qs < 2; ++qs) {
#pragma unroll
      for (int j = 0; j < 8; ++j) {
        const int row = m0 + 16 * qs + j + 8 * h;
        float v = (acc[qs][ns][j] + bb) * scale;
        if (mode == 2)
          out[(size_t)head * 64 * 4096 + (size_t)dcol * 4096 + row] = (half_t)v;
        else
          out[(size_t)head * 4096 * 64 + (size_t)row * 64 + dcol] = (half_t)v;
      }
    }
  }
}

// ---------------------------------------------------------------------------
// Flash attention: one 32-row Q tile per wave (2 sub-tiles), kv in blocks of 32.
// Every K/V fragment feeds 2 WMMAs. Per-wave private LDS bounce tile for P;
// no barriers needed: LDS ops from one wave are in-order (ISA 7.3) and the
// compiler tracks the aliasing store->load through the shared array.
// Q pre-scaled by 1/sqrt(64). Writes ctx as f16 [L][1024] (head-concatenated).
__global__ __launch_bounds__(128) void flash_attn_kernel(
    const half_t* __restrict__ Q, const half_t* __restrict__ K,
    const half_t* __restrict__ Vt, half_t* __restrict__ ctx) {
  __shared__ __align__(16) half_t plds[4][2][16 * 32];  // 4 waves x 2 P tiles
  const int wave = threadIdx.x >> 5;
  const int gid  = blockIdx.x * 4 + wave;      // 2048 = 16 heads * 128 qtiles
  const int head = gid >> 7;
  const int q0   = (gid & 127) * 32;
  const int lane = threadIdx.x & 31;
  const int col  = lane & 15, h = lane >> 4;

  const half_t* Qh = Q + (size_t)head * 4096 * 64;
  const half_t* Kh = K + (size_t)head * 4096 * 64;
  const half_t* Vh = Vt + (size_t)head * 64 * 4096;

  v16h aq[2][2];
#pragma unroll
  for (int qs = 0; qs < 2; ++qs) {
    aq[qs][0] = load_frag_a(Qh + (size_t)(q0 + 16 * qs) * 64 + 0, 64);
    aq[qs][1] = load_frag_a(Qh + (size_t)(q0 + 16 * qs) * 64 + 32, 64);
  }

  float m[2][8], l[2][8];
  v8f acc[2][4] = {};
#pragma unroll
  for (int qs = 0; qs < 2; ++qs)
#pragma unroll
    for (int j = 0; j < 8; ++j) { m[qs][j] = -1e30f; l[qs][j] = 0.0f; }

  for (int kb = 0; kb < 4096; kb += 32) {
    __builtin_prefetch(Kh + (size_t)(kb + 32) * 64, 0, 0);
    // S = Q K^T : 4 shared K fragments -> 8 WMMAs (each fragment used twice)
    v16h bk[2][2];
#pragma unroll
    for (int t = 0; t < 2; ++t) {
      bk[t][0] = load_frag_b(Kh + (size_t)(kb + 16 * t) * 64 + 0, 64);
      bk[t][1] = load_frag_b(Kh + (size_t)(kb + 16 * t) * 64 + 32, 64);
    }
    v8f s[2][2];
#pragma unroll
    for (int qs = 0; qs < 2; ++qs)
#pragma unroll
      for (int t = 0; t < 2; ++t) {
        v8f z = {};
        z = wmma_f32(aq[qs][0], bk[t][0], z);
        z = wmma_f32(aq[qs][1], bk[t][1], z);
        s[qs][t] = z;
      }
    // Online softmax per 16-row sub-tile. C-layout: element j -> row j+8h, so
    // row stats are 16-lane butterflies (xor 1,2,4,8 stays within the half).
#pragma unroll
    for (int qs = 0; qs < 2; ++qs) {
#pragma unroll
      for (int j = 0; j < 8; ++j) {
        float rm = fmaxf(s[qs][0][j], s[qs][1][j]);
        rm = fmaxf(rm, __shfl_xor(rm, 1, 32));
        rm = fmaxf(rm, __shfl_xor(rm, 2, 32));
        rm = fmaxf(rm, __shfl_xor(rm, 4, 32));
        rm = fmaxf(rm, __shfl_xor(rm, 8, 32));
        const float mn = fmaxf(m[qs][j], rm);
        const float cf = __expf(m[qs][j] - mn);
        const float p0 = __expf(s[qs][0][j] - mn);
        const float p1 = __expf(s[qs][1][j] - mn);
        float rs = p0 + p1;
        rs += __shfl_xor(rs, 1, 32);
        rs += __shfl_xor(rs, 2, 32);
        rs += __shfl_xor(rs, 4, 32);
        rs += __shfl_xor(rs, 8, 32);
        l[qs][j] = l[qs][j] * cf + rs;
        m[qs][j] = mn;
        s[qs][0][j] = p0; s[qs][1][j] = p1;
#pragma unroll
        for (int t = 0; t < 4; ++t) acc[qs][t][j] *= cf;
      }
      // Re-lay P (C layout) -> A-fragment layout via per-wave LDS tile
      half_t* P = &plds[wave][qs][0];
#pragma unroll
      for (int j = 0; j < 8; ++j) {
        const int row = j + 8 * h;
        P[row * 32 + col]      = (half_t)s[qs][0][j];
        P[row * 32 + 16 + col] = (half_t)s[qs][1][j];
      }
    }
    const v16h ap0 = load_frag_a(&plds[wave][0][0], 32);
    const v16h ap1 = load_frag_a(&plds[wave][1][0], 32);
    // ctx += P @ V : 4 shared V fragments -> 8 WMMAs (each fragment used twice)
#pragma unroll
    for (int t = 0; t < 4; ++t) {
      v16h bv = load_frag_b(Vh + (size_t)(t * 16) * 4096 + kb, 4096);
      acc[0][t] = wmma_f32(ap0, bv, acc[0][t]);
      acc[1][t] = wmma_f32(ap1, bv, acc[1][t]);
    }
  }

#pragma unroll
  for (int qs = 0; qs < 2; ++qs)
#pragma unroll
    for (int j = 0; j < 8; ++j) {
      const float inv = 1.0f / l[qs][j];
      const int row = q0 + 16 * qs + j + 8 * h;
#pragma unroll
      for (int t = 0; t < 4; ++t) {
        ctx[(size_t)row * 1024 + head * 64 + t * 16 + col] =
            (half_t)(acc[qs][t][j] * inv);
      }
    }
}

// ---------------------------------------------------------------------------
// Output projection: out(f32) = ctx16 @ Wo^T + bo, 32x32 tile per wave.
__global__ __launch_bounds__(128) void out_proj_kernel(
    const half_t* __restrict__ ctx, const half_t* __restrict__ wot,
    const float* __restrict__ bo, float* __restrict__ out) {
  const int wave = threadIdx.x >> 5;
  const int gid  = blockIdx.x * 4 + wave;
  const int m0   = (gid >> 5) * 32;
  const int n0   = (gid & 31) * 32;
  const int lane = threadIdx.x & 31;
  const int col  = lane & 15, h = lane >> 4;

  v8f acc[2][2] = {};
  const half_t* arow0 = ctx + (size_t)m0 * 1024;
  const half_t* arow1 = ctx + (size_t)(m0 + 16) * 1024;
  const half_t* brow0 = wot + (size_t)n0 * 1024;
  const half_t* brow1 = wot + (size_t)(n0 + 16) * 1024;
  for (int k = 0; k < 1024; k += 32) {
    v16h a0 = load_frag_a(arow0 + k, 1024);
    v16h a1 = load_frag_a(arow1 + k, 1024);
    v16h b0 = load_frag_b(brow0 + k, 1024);
    v16h b1 = load_frag_b(brow1 + k, 1024);
    acc[0][0] = wmma_f32(a0, b0, acc[0][0]);
    acc[0][1] = wmma_f32(a0, b1, acc[0][1]);
    acc[1][0] = wmma_f32(a1, b0, acc[1][0]);
    acc[1][1] = wmma_f32(a1, b1, acc[1][1]);
  }
#pragma unroll
  for (int ns = 0; ns < 2; ++ns) {
    const float b = bo[n0 + 16 * ns + col];
#pragma unroll
    for (int qs = 0; qs < 2; ++qs)
#pragma unroll
      for (int j = 0; j < 8; ++j) {
        out[(size_t)(m0 + 16 * qs + j + 8 * h) * 1024 + n0 + 16 * ns + col] =
            acc[qs][ns][j] + b;
      }
  }
}

// ---------------------------------------------------------------------------
extern "C" void kernel_launch(void* const* d_in, const int* in_sizes, int n_in,
                              void* d_out, int out_size, void* d_ws, size_t ws_size,
                              hipStream_t stream) {
  const float* x  = (const float*)d_in[0];
  const float* Wq = (const float*)d_in[1];
  const float* bq = (const float*)d_in[2];
  const float* Wk = (const float*)d_in[3];
  const float* bk = (const float*)d_in[4];
  const float* Wv = (const float*)d_in[5];
  const float* bv = (const float*)d_in[6];
  const float* Wo = (const float*)d_in[7];
  const float* bo = (const float*)d_in[8];
  float* out = (float*)d_out;

  char* ws = (char*)d_ws;
  const size_t MB = 1ull << 20;
  half_t* x16  = (half_t*)(ws + 0 * MB);   // 8 MB  : x as f16 [L][D]
  half_t* wqt  = (half_t*)(ws + 8 * MB);   // 2 MB  : Wq^T f16 [N][K]
  half_t* wkt  = (half_t*)(ws + 10 * MB);  // 2 MB
  half_t* wvt  = (half_t*)(ws + 12 * MB);  // 2 MB
  half_t* wot  = (half_t*)(ws + 14 * MB);  // 2 MB
  half_t* Q16  = (half_t*)(ws + 16 * MB);  // 8 MB  : [H][L][64], pre-scaled 1/8
  half_t* K16  = (half_t*)(ws + 24 * MB);  // 8 MB  : [H][L][64]
  half_t* V16t = (half_t*)(ws + 32 * MB);  // 8 MB  : [H][64][L]  (transposed)
  half_t* c16  = (half_t*)(ws + 40 * MB);  // 8 MB  : ctx f16 [L][D]

  cvt_f32_f16<<<16384, 256, 0, stream>>>(x, x16, 4096 * 1024);
  cvt_transpose<<<4096, 256, 0, stream>>>(Wq, wqt);
  cvt_transpose<<<4096, 256, 0, stream>>>(Wk, wkt);
  cvt_transpose<<<4096, 256, 0, stream>>>(Wv, wvt);
  cvt_transpose<<<4096, 256, 0, stream>>>(Wo, wot);

  qkv_proj_kernel<<<1024, 128, 0, stream>>>(x16, wqt, bq, Q16, 0, 0.125f);
  qkv_proj_kernel<<<1024, 128, 0, stream>>>(x16, wkt, bk, K16, 1, 1.0f);
  qkv_proj_kernel<<<1024, 128, 0, stream>>>(x16, wvt, bv, V16t, 2, 1.0f);

  flash_attn_kernel<<<512, 128, 0, stream>>>(Q16, K16, V16t, c16);

  out_proj_kernel<<<1024, 128, 0, stream>>>(c16, wot, bo, out);
}
